// AttentiveRelationalModuleUniformObs_13838384628346
// MI455X (gfx1250) — matrix operand order repaired and checked by reading, not compiled, not run on hardware
//
#include <hip/hip_runtime.h>
#include <hip/hip_bf16.h>

// AttentiveRelationalModule for MI455X (gfx1250, wave32, WMMA).
//
// Math note: softmax over the neighbor axis cancels any term constant in n.
// The self-embedding block, neighbor-mean block and bias ba of the attention
// linear are all constant over n, so att = softmax_n(relu(X@Wc+bc) @ Wa[EC:2EC,:]).
// local_data and ba are therefore (exactly) unused.

typedef _Float16 f16;
typedef __attribute__((ext_vector_type(16))) _Float16 v16h;
typedef __attribute__((ext_vector_type(8)))  float    v8f;

#define BATCH 1024
#define NB    256   // neighbors
#define DD    64    // input dim
#define LATD  128   // latent dim
#define ECD   32    // comm embedding dim

// ---- LDS layout (bytes) ----
#define XH_OFF   0                              // Xh  [256][64]  f16 : 32768
#define WFT_OFF  (XH_OFF  + NB*DD*2)            // WfT [128][64]  f16 : 16384
#define WCT_OFF  (WFT_OFF + LATD*DD*2)          // WcT [32][64]   f16 : 4096
#define WAT_OFF  (WCT_OFF + ECD*DD*2)           // WaT [128][32]  f16 : 8192
#define FH_OFF   (WAT_OFF + LATD*ECD*2)         // Fh  [256][128] f16 : 65536
#define CH_OFF   (FH_OFF  + NB*LATD*2)          // Ch  [256][32]  f16 : 16384
#define LF_OFF   (CH_OFF  + NB*ECD*2)           // Lf  [256][128] f32 : 131072
#define BFS_OFF  (LF_OFF  + NB*LATD*4)          // bf  [128] f32
#define BCS_OFF  (BFS_OFF + LATD*4)             // bc  [32]  f32
#define PM_OFF   (BCS_OFF + ECD*4)              // partial max [256]
#define PS_OFF   (PM_OFF  + NB*4)               // partial sum [256]
#define PA_OFF   (PS_OFF  + NB*4)               // partial acc [256]
#define AGG_OFF  (PA_OFF  + NB*4)               // agg [128] f32
#define SMEM_BYTES (AGG_OFF + LATD*4)           // = 278656 bytes (< 320KB WGP LDS)

union U16h { v16h v; f16 e[16]; };

// A-fragment (16x32 f16, row-major source with row stride ld halves).
// ISA layout: lanes0-15/16-31 both hold M=lane&15; halves 0..7 -> K=kk0..kk0+7,
// halves 8..15 -> K=kk0+16..kk0+23, kk0 = 8*(lane>=16) + 32*kb.
__device__ __forceinline__ v16h frag_a16(const f16* rowp, int hi, int kb) {
    U16h u;
    const int base = kb * 32 + hi * 8;
#pragma unroll
    for (int i = 0; i < 8; ++i) {
        u.e[i]     = rowp[base + i];
        u.e[8 + i] = rowp[base + 16 + i];
    }
    return u.v;
}

// B-fragment (32x16 f16) from a transposed (column-major) weight array:
// colp points at column n = n0 + (lane&15); half h<8 -> K = 8*(lane>=16) + h + 32*kb,
// h>=8 -> K = 16 + 8*(lane>=16) + (h-8) + 32*kb.
__device__ __forceinline__ v16h frag_b16(const f16* colp, int hi, int kb) {
    U16h u;
    const int base = kb * 32 + hi * 8;
#pragma unroll
    for (int i = 0; i < 8; ++i) {
        u.e[i]     = colp[base + i];
        u.e[8 + i] = colp[base + 16 + i];
    }
    return u.v;
}

__global__ __launch_bounds__(256) void arm_fused_kernel(
    const float* __restrict__ neighbor,  // [B,N,D]
    const float* __restrict__ Wc,        // [D,EC]
    const float* __restrict__ bc,        // [EC]
    const float* __restrict__ Wf,        // [D,LAT]
    const float* __restrict__ bf,        // [LAT]
    const float* __restrict__ Wa,        // [3*EC,LAT]
    const float* __restrict__ Wl,        // [LAT,LAT]
    const float* __restrict__ bl,        // [LAT]
    float* __restrict__ out)             // [B,LAT]
{
    extern __shared__ char smem[];
    f16*   Xh  = (f16*)  (smem + XH_OFF);
    f16*   WfT = (f16*)  (smem + WFT_OFF);
    f16*   WcT = (f16*)  (smem + WCT_OFF);
    f16*   WaT = (f16*)  (smem + WAT_OFF);
    f16*   Fh  = (f16*)  (smem + FH_OFF);
    f16*   Ch  = (f16*)  (smem + CH_OFF);
    float* Lf  = (float*)(smem + LF_OFF);
    float* bfs = (float*)(smem + BFS_OFF);
    float* bcs = (float*)(smem + BCS_OFF);
    float* pm  = (float*)(smem + PM_OFF);
    float* ps  = (float*)(smem + PS_OFF);
    float* pa  = (float*)(smem + PA_OFF);
    float* agg = (float*)(smem + AGG_OFF);

    const int tid  = threadIdx.x;
    const int lane = tid & 31;
    const int wave = tid >> 5;         // 8 waves
    const int b    = blockIdx.x;
    const int hi   = lane >> 4;        // half-wave select
    const int lj   = lane & 15;

    // ---------- Stage 0: stage X + weights into LDS (f32 -> f16) ----------
    {
        const float2* src = (const float2*)(neighbor + (size_t)b * NB * DD);
        unsigned* xhp = (unsigned*)Xh;
#pragma unroll 4
        for (int i = tid; i < NB * DD / 2; i += 256) {   // coalesced 8B loads
            float2 v = src[i];
            union { f16 h[2]; unsigned u; } p;
            p.h[0] = (f16)v.x; p.h[1] = (f16)v.y;
            xhp[i] = p.u;
        }
        // Transposed weights: WT[n][k] = W[k][n]; k fast -> contiguous LDS writes.
        for (int i = tid; i < DD * LATD; i += 256) {
            int k = i & (DD - 1), n = i >> 6;
            WfT[n * DD + k] = (f16)Wf[k * LATD + n];
        }
        for (int i = tid; i < DD * ECD; i += 256) {
            int k = i & (DD - 1), n = i >> 6;
            WcT[n * DD + k] = (f16)Wc[k * ECD + n];
        }
        // Middle block of Wa (rows EC..2EC-1): WaT[j][k] = Wa[EC+k][j]
        for (int i = tid; i < ECD * LATD; i += 256) {
            int k = i & (ECD - 1), j = i >> 5;
            WaT[j * ECD + k] = (f16)Wa[(ECD + k) * LATD + j];
        }
        if (tid < LATD) bfs[tid] = bf[tid];
        if (tid < ECD)  bcs[tid] = bc[tid];
    }
    __syncthreads();

    // ---------- Stage 1: F = relu(X@Wf+bf) -> Fh ; C = relu(X@Wc+bc) -> Ch ----------
#pragma unroll
    for (int t = 0; t < 2; ++t) {
        const int m0 = (wave * 2 + t) * 16;
        const f16* arow = Xh + (m0 + lj) * DD;
        v16h a0 = frag_a16(arow, hi, 0);
        v16h a1 = frag_a16(arow, hi, 1);
        const int mb = m0 + hi * 8;

        // F: 8 N-tiles, K=64 (2 wmma each)
        for (int nt = 0; nt < LATD / 16; ++nt) {
            const int n0 = nt * 16;
            const f16* bcol = WfT + (n0 + lj) * DD;
            v16h b0 = frag_b16(bcol, hi, 0);
            v16h b1 = frag_b16(bcol, hi, 1);
            v8f acc = {};
            acc = __builtin_amdgcn_wmma_f32_16x16x32_f16(false, a0, false, b0, (short)0, acc, false, false);
            acc = __builtin_amdgcn_wmma_f32_16x16x32_f16(false, a1, false, b1, (short)0, acc, false, false);
            const int n = n0 + lj;
            const float bv = bfs[n];
#pragma unroll
            for (int r = 0; r < 8; ++r) {
                float v = acc[r] + bv;
                v = v > 0.f ? v : 0.f;
                Fh[(mb + r) * LATD + n] = (f16)v;
            }
        }
        // C: 2 N-tiles, K=64
        for (int nt = 0; nt < ECD / 16; ++nt) {
            const int n0 = nt * 16;
            const f16* bcol = WcT + (n0 + lj) * DD;
            v16h b0 = frag_b16(bcol, hi, 0);
            v16h b1 = frag_b16(bcol, hi, 1);
            v8f acc = {};
            acc = __builtin_amdgcn_wmma_f32_16x16x32_f16(false, a0, false, b0, (short)0, acc, false, false);
            acc = __builtin_amdgcn_wmma_f32_16x16x32_f16(false, a1, false, b1, (short)0, acc, false, false);
            const int n = n0 + lj;
            const float bv = bcs[n];
#pragma unroll
            for (int r = 0; r < 8; ++r) {
                float v = acc[r] + bv;
                v = v > 0.f ? v : 0.f;
                Ch[(mb + r) * ECD + n] = (f16)v;
            }
        }
    }
    __syncthreads();

    // ---------- Stage 2: attention logits L = Ch @ Wa_mid (K=32) -> Lf (f32) ----------
#pragma unroll
    for (int t = 0; t < 2; ++t) {
        const int m0 = (wave * 2 + t) * 16;
        const f16* arow = Ch + (m0 + lj) * ECD;
        v16h a = frag_a16(arow, hi, 0);
        const int mb = m0 + hi * 8;
        for (int nt = 0; nt < LATD / 16; ++nt) {
            const int n0 = nt * 16;
            const f16* bcol = WaT + (n0 + lj) * ECD;
            v16h bm = frag_b16(bcol, hi, 0);
            v8f acc = {};
            acc = __builtin_amdgcn_wmma_f32_16x16x32_f16(false, a, false, bm, (short)0, acc, false, false);
            const int n = n0 + lj;
#pragma unroll
            for (int r = 0; r < 8; ++r) Lf[(mb + r) * LATD + n] = acc[r];
        }
    }
    __syncthreads();

    // ---------- Stage 3: column-wise online softmax + weighted sum over n ----------
    {
        const int k  = tid & (LATD - 1);
        const int hh = tid >> 7;            // 0/1 : half of the neighbor range
        float m = -3.4e38f, s = 0.f, a = 0.f;
        const int n0 = hh * 128;
        for (int n = n0; n < n0 + 128; ++n) {
            float x  = Lf[n * LATD + k];
            float f  = (float)Fh[n * LATD + k];
            float mn = fmaxf(m, x);
            float c  = __expf(m - mn);
            float e  = __expf(x - mn);
            s = s * c + e;
            a = a * c + e * f;
            m = mn;
        }
        pm[tid] = m; ps[tid] = s; pa[tid] = a;
    }
    __syncthreads();
    if (tid < LATD) {
        float m0 = pm[tid], m1 = pm[tid + 128];
        float M  = fmaxf(m0, m1);
        float c0 = __expf(m0 - M), c1 = __expf(m1 - M);
        float s  = ps[tid] * c0 + ps[tid + 128] * c1;
        float a  = pa[tid] * c0 + pa[tid + 128] * c1;
        agg[tid] = a / s;
    }
    __syncthreads();

    // ---------- Stage 4: out = relu(agg @ Wl + bl), Wl reads are L2-resident ----------
    if (tid < LATD) {
        float acc = bl[tid];
#pragma unroll 4
        for (int k = 0; k < LATD; ++k)
            acc += agg[k] * Wl[k * LATD + tid];   // coalesced across threads
        out[(size_t)b * LATD + tid] = acc > 0.f ? acc : 0.f;
    }
}

extern "C" void kernel_launch(void* const* d_in, const int* in_sizes, int n_in,
                              void* d_out, int out_size, void* d_ws, size_t ws_size,
                              hipStream_t stream) {
    (void)in_sizes; (void)n_in; (void)out_size; (void)d_ws; (void)ws_size;
    // setup_inputs order: local_data, neighbor_data, Wc, bc, Wf, bf, Wa, ba, Wl, bl
    // local_data (d_in[0]) and ba (d_in[7]) cancel exactly inside the softmax.
    const float* neighbor = (const float*)d_in[1];
    const float* Wc       = (const float*)d_in[2];
    const float* bc       = (const float*)d_in[3];
    const float* Wf       = (const float*)d_in[4];
    const float* bf       = (const float*)d_in[5];
    const float* Wa       = (const float*)d_in[6];
    const float* Wl       = (const float*)d_in[8];
    const float* bl       = (const float*)d_in[9];
    float* out            = (float*)d_out;

    dim3 grid(BATCH), block(256);
    hipLaunchKernelGGL(arm_fused_kernel, grid, block, SMEM_BYTES, stream,
                       neighbor, Wc, bc, Wf, bf, Wa, Wl, bl, out);
}
